// WindowAttention_8134668059220
// MI455X (gfx1250) — matrix operand order repaired and checked by reading, not compiled, not run on hardware
//
#include <hip/hip_runtime.h>
#include <hip/hip_bf16.h>

typedef __attribute__((ext_vector_type(16))) _Float16 v16h;
typedef __attribute__((ext_vector_type(8)))  _Float16 v8h;
typedef __attribute__((ext_vector_type(8)))  float    v8f;

#define WIN_B    4096
#define WIN_N    49
#define WIN_NP   64      // padded tokens
#define WIN_C    128
#define WIN_H    4
#define WIN_D    32
#define WIN_3C   384
#define Q_SCALE  0.17677669529663687f   // 32^-0.5

// ---- LDS layout offsets (bytes), all 16B aligned ----
#define OFF_W    0                       // f16 [384][128] qkv_w^T (reused: [128][128] proj_w^T) : 98304
#define OFF_X    98304                   // f16 [64][128] x (reused as attn-output staging)       : 16384
#define OFF_Q    114688                  // f16 [4][64][32] scaled q                              : 16384
#define OFF_K    131072                  // f16 [4][64][32] k (natural == B^T for q@k^T)          : 16384
#define OFF_VT   147456                  // f16 [4][32][64] v^T (B^T for attn@v)                  : 16384
#define OFF_S    163840                  // f32 [4][64][64] raw scores                            : 65536
#define OFF_P    229376                  // f16 [4][64][64] softmax probs                         : 32768
#define OFF_BIAS 262144                  // f32 [384] qkv_b (reused: [128] proj_b)                : 1536
#define SMEM_BYTES 263680

// A-fragment (16x32 f16, M x K): lanes 0-15 -> M=lane, K in {0..7,16..23};
// lanes 16-31 -> M=lane-16, K in {8..15,24..31}. Two 16B contiguous LDS loads.
static __device__ inline v16h load_frag_a(const _Float16* base, int ld, int row0, int k0, int lane) {
    const int r  = row0 + (lane & 15);
    const int kb = k0 + ((lane >> 4) << 3);
    const _Float16* p = base + r * ld + kb;
    v8h lo = *(const v8h*)(p);
    v8h hi = *(const v8h*)(p + 16);
    v16h a;
#pragma unroll
    for (int i = 0; i < 8; ++i) { a[i] = lo[i]; a[8 + i] = hi[i]; }
    return a;
}

// B-fragment (32x16 f16, K x N) loaded from B^T storage [N][K]:
// lanes 0-15 -> N=lane, K in {0..7,16..23}; lanes 16-31 -> N=lane-16, K in {8..15,24..31}.
// Two 16B contiguous reads per lane.
static __device__ inline v16h load_frag_b(const _Float16* baseT, int ldk, int col0, int k0, int lane) {
    const int n  = col0 + (lane & 15);
    const int kb = k0 + ((lane >> 4) << 3);
    const _Float16* p = baseT + n * ldk + kb;
    v8h lo = *(const v8h*)(p);
    v8h hi = *(const v8h*)(p + 16);
    v16h b;
#pragma unroll
    for (int i = 0; i < 8; ++i) { b[i] = lo[i]; b[8 + i] = hi[i]; }
    return b;
}

static __device__ inline v8f wmma_f16(v16h a, v16h b, v8f c) {
    return __builtin_amdgcn_wmma_f32_16x16x32_f16(false, a, false, b, (short)0, c, false, false);
}

__global__ __launch_bounds__(256) void win_attn_kernel(
    const float* __restrict__ x,        const float* __restrict__ mask,
    const float* __restrict__ qkv_w,    const float* __restrict__ qkv_b,
    const float* __restrict__ proj_w,   const float* __restrict__ proj_b,
    const float* __restrict__ bias_tab, const int*  __restrict__ rel_index,
    float* __restrict__ out)
{
    extern __shared__ __align__(16) char smem[];
    _Float16* sW   = (_Float16*)(smem + OFF_W);
    _Float16* sX   = (_Float16*)(smem + OFF_X);
    _Float16* sQ   = (_Float16*)(smem + OFF_Q);
    _Float16* sK   = (_Float16*)(smem + OFF_K);
    _Float16* sVT  = (_Float16*)(smem + OFF_VT);
    float*    sS   = (float*)   (smem + OFF_S);
    _Float16* sP   = (_Float16*)(smem + OFF_P);
    float*    sBias= (float*)   (smem + OFF_BIAS);

    const int b    = blockIdx.x;
    const int tid  = threadIdx.x;
    const int lane = tid & 31;
    const int wave = tid >> 5;     // 0..7
    const int nloc = lane & 15;    // fragment N coordinate
    const int moff = (lane >> 4) << 3;   // fragment M base (0 or 8)

    const size_t mask_base = (size_t)b * WIN_N * WIN_N;
    __builtin_prefetch(mask + mask_base, 0, 0);   // global_prefetch_b8

    // ---- Stage 1: qkv_w^T -> LDS f16, qkv_b, x (padded) -> LDS f16 ----
    for (int i = tid; i < WIN_C * WIN_3C; i += 256) {
        int k = i / WIN_3C, n = i % WIN_3C;       // global read coalesced over n
        sW[n * WIN_C + k] = (_Float16)qkv_w[i];
    }
    for (int i = tid; i < WIN_3C; i += 256) sBias[i] = qkv_b[i];
    for (int i = tid; i < WIN_NP * WIN_C; i += 256) {
        int r = i >> 7, c = i & 127;
        sX[i] = (r < WIN_N) ? (_Float16)x[(size_t)b * WIN_N * WIN_C + r * WIN_C + c]
                            : (_Float16)0.f;
    }
    __syncthreads();

    // ---- Stage 2: QKV GEMM [64x128]x[128x384], 4-wide N register blocking.
    // 24 groups: g = mt*6 + ng, group ng covers nt = ng*4..ng*4+3.
    // Region is uniform per group: ng 0-1 -> q, 2-3 -> k, 4-5 -> v.
    for (int g = wave; g < 24; g += 8) {
        const int mt = g / 6, ng = g % 6;
        v8f acc[4] = {{}, {}, {}, {}};
#pragma unroll
        for (int kk = 0; kk < WIN_C; kk += 32) {
            v16h af = load_frag_a(sX, WIN_C, mt * 16, kk, lane);
#pragma unroll
            for (int j = 0; j < 4; ++j) {
                v16h bf = load_frag_b(sW, WIN_C, (ng * 4 + j) * 16, kk, lane);
                acc[j] = wmma_f16(af, bf, acc[j]);
            }
        }
        const int m0 = mt * 16 + moff;
        if (ng < 2) {                               // ---- q (scaled) ----
#pragma unroll
            for (int j = 0; j < 4; ++j) {
                const int c = (ng * 4 + j) * 16 + nloc;      // 0..127
                const int h = c >> 5, d = c & 31;
                const float bias = sBias[c];
#pragma unroll
                for (int r = 0; r < 8; ++r)
                    sQ[(h * WIN_NP + m0 + r) * WIN_D + d] =
                        (_Float16)((acc[j][r] + bias) * Q_SCALE);
            }
        } else if (ng < 4) {                        // ---- k (natural == B^T) ----
#pragma unroll
            for (int j = 0; j < 4; ++j) {
                const int n = (ng * 4 + j) * 16 + nloc;      // 128..255
                const int c = n - 128, h = c >> 5, d = c & 31;
                const float bias = sBias[n];
#pragma unroll
                for (int r = 0; r < 8; ++r)
                    sK[(h * WIN_NP + m0 + r) * WIN_D + d] = (_Float16)(acc[j][r] + bias);
            }
        } else {                                    // ---- v (transposed) ----
#pragma unroll
            for (int j = 0; j < 4; ++j) {
                const int n = (ng * 4 + j) * 16 + nloc;      // 256..383
                const int c = n - 256, h = c >> 5, d = c & 31;
                const float bias = sBias[n];
#pragma unroll
                for (int r = 0; r < 8; ++r)
                    sVT[(h * WIN_D + d) * WIN_NP + m0 + r] = (_Float16)(acc[j][r] + bias);
            }
        }
    }
    __syncthreads();

    // ---- Stage 3: scores S = q @ k^T per head (K=32); 4 N-tiles per A fragment ----
    for (int g = wave; g < 16; g += 8) {            // g = h*4 + mt
        const int h = g >> 2, mt = g & 3;
        v16h af = load_frag_a(sQ + h * WIN_NP * WIN_D, WIN_D, mt * 16, 0, lane);
        v8f acc[4];
#pragma unroll
        for (int j = 0; j < 4; ++j) {
            v16h bf = load_frag_b(sK + h * WIN_NP * WIN_D, WIN_D, j * 16, 0, lane);
            v8f z = {};
            acc[j] = wmma_f16(af, bf, z);
        }
        const int m0 = mt * 16 + moff;
#pragma unroll
        for (int j = 0; j < 4; ++j) {
            const int n = j * 16 + nloc;
#pragma unroll
            for (int r = 0; r < 8; ++r)
                sS[(h * WIN_NP + m0 + r) * WIN_NP + n] = acc[j][r];
        }
    }
    __syncthreads();

    // ---- Stage 4: softmax (+rel-pos bias gather, +window mask); 1 thread = 1 (head,row) ----
    {
        const int h = tid >> 6, r = tid & 63;
        _Float16* prow = sP + (h * WIN_NP + r) * WIN_NP;
        if (r < WIN_N) {
            const float* srow = sS + (h * WIN_NP + r) * WIN_NP;
            const float* mrow = mask + mask_base + (size_t)r * WIN_N;
            const int*   irow = rel_index + r * WIN_N;
            float maxv = -3.0e38f;
            for (int j = 0; j < WIN_N; ++j) {
                float vv = srow[j] + bias_tab[irow[j] * WIN_H + h] + mrow[j];
                maxv = fmaxf(maxv, vv);
            }
            float sum = 0.f;
            for (int j = 0; j < WIN_N; ++j) {
                float vv = srow[j] + bias_tab[irow[j] * WIN_H + h] + mrow[j];
                float e = __expf(vv - maxv);
                sum += e;
                prow[j] = (_Float16)e;               // unnormalized
            }
            const float inv = 1.f / sum;
            for (int j = 0; j < WIN_N; ++j)
                prow[j] = (_Float16)((float)prow[j] * inv);
            for (int j = WIN_N; j < WIN_NP; ++j) prow[j] = (_Float16)0.f;
        } else {
            // padded rows feed only discarded output rows; write a finite delta row
            prow[0] = (_Float16)1.f;
            for (int j = 1; j < WIN_NP; ++j) prow[j] = (_Float16)0.f;
        }
    }
    __syncthreads();

    // ---- Stage 5: stage proj_w^T + proj_b (sW free now), and out = P @ v ----
    for (int i = tid; i < WIN_C * WIN_C; i += 256) {
        int k = i >> 7, n = i & 127;
        sW[n * WIN_C + k] = (_Float16)proj_w[i];
    }
    for (int i = tid; i < WIN_C; i += 256) sBias[i] = proj_b[i];

    for (int g = wave; g < 16; g += 8) {            // g = h*4 + mt ; both N-tiles per group
        const int h = g >> 2, mt = g & 3;
        v8f acc[2] = {{}, {}};
#pragma unroll
        for (int kk = 0; kk < WIN_NP; kk += 32) {
            v16h af = load_frag_a(sP + h * WIN_NP * WIN_NP, WIN_NP, mt * 16, kk, lane);
#pragma unroll
            for (int j = 0; j < 2; ++j) {
                v16h bf = load_frag_b(sVT + h * WIN_D * WIN_NP, WIN_NP, j * 16, kk, lane);
                acc[j] = wmma_f16(af, bf, acc[j]);
            }
        }
        const int m0 = mt * 16 + moff;
#pragma unroll
        for (int j = 0; j < 2; ++j) {
            const int n = h * WIN_D + j * 16 + nloc;         // channel = h*32 + d
#pragma unroll
            for (int r = 0; r < 8; ++r)
                sX[(m0 + r) * WIN_C + n] = (_Float16)acc[j][r];   // sX reused as staging
        }
    }
    __syncthreads();

    // ---- Stage 6: projection GEMM [64x128]x[128x128] + bias -> global (rows < 49) ----
    {                                                // 8 groups, exactly one per wave
        const int mt = wave >> 1, ng = wave & 1;     // nt = ng*4 .. ng*4+3
        v8f acc[4] = {{}, {}, {}, {}};
#pragma unroll
        for (int kk = 0; kk < WIN_C; kk += 32) {
            v16h af = load_frag_a(sX, WIN_C, mt * 16, kk, lane);
#pragma unroll
            for (int j = 0; j < 4; ++j) {
                v16h bf = load_frag_b(sW, WIN_C, (ng * 4 + j) * 16, kk, lane);
                acc[j] = wmma_f16(af, bf, acc[j]);
            }
        }
        const int m0 = mt * 16 + moff;
#pragma unroll
        for (int j = 0; j < 4; ++j) {
            const int n = (ng * 4 + j) * 16 + nloc;
            const float bias = sBias[n];
#pragma unroll
            for (int r = 0; r < 8; ++r) {
                const int m = m0 + r;
                if (m < WIN_N)
                    out[(size_t)b * WIN_N * WIN_C + (size_t)m * WIN_C + n] = acc[j][r] + bias;
            }
        }
    }
}

extern "C" void kernel_launch(void* const* d_in, const int* in_sizes, int n_in,
                              void* d_out, int out_size, void* d_ws, size_t ws_size,
                              hipStream_t stream) {
    (void)in_sizes; (void)n_in; (void)out_size; (void)d_ws; (void)ws_size;
    const float* x        = (const float*)d_in[0];
    const float* mask     = (const float*)d_in[1];
    const float* qkv_w    = (const float*)d_in[2];
    const float* qkv_b    = (const float*)d_in[3];
    const float* proj_w   = (const float*)d_in[4];
    const float* proj_b   = (const float*)d_in[5];
    const float* bias_tab = (const float*)d_in[6];
    const int*   rel_idx  = (const int*)d_in[7];
    float* out = (float*)d_out;

    // Allow >64KB dynamic LDS (gfx1250 WGP has 320KB). Deterministic, capture-safe.
    hipFuncSetAttribute((const void*)win_attn_kernel,
                        hipFuncAttributeMaxDynamicSharedMemorySize, SMEM_BYTES);

    win_attn_kernel<<<WIN_B, 256, SMEM_BYTES, stream>>>(
        x, mask, qkv_w, qkv_b, proj_w, proj_b, bias_tab, rel_idx, out);
}
